// Model_39676907887947
// MI455X (gfx1250) — matrix operand order repaired and checked by reading, not compiled, not run on hardware
//
#include <hip/hip_runtime.h>

// Flash attention forward, (B=8, H=16, N=1024, D=64) fp32 in/out, gfx1250.
// Block = 128 threads = 4 waves; each wave owns a 16-row Q tile of one (b,h).
// Per 64-key superblock: block cooperatively stages K (row-major) and V
// (transposed) as f16 into double-buffered LDS while computing on the
// previous tile (one barrier/iter, global-load latency hidden behind the
// 18-WMMA compute phase). Online softmax in exp2 domain: DPP-fused
// v_max_num butterflies for row-max, P@ones WMMAs for row-sum, wave-uniform
// skip of the O rescale when no row max grew.

typedef __attribute__((ext_vector_type(16))) _Float16 v16h;
typedef __attribute__((ext_vector_type(8)))  _Float16 v8h;
typedef __attribute__((ext_vector_type(4)))  _Float16 v4h;
typedef __attribute__((ext_vector_type(8)))  float    v8f;
typedef __attribute__((ext_vector_type(4)))  float    v4f;

#define SEQ 1024
#define HD  64
#define NTILES (SEQ / 64)
#define KSTRIDE 72   // padded row stride (halves): 16 frag lanes -> 16 banks
#define VSTRIDE 72
#define PSTRIDE 72

// Raw IEEE maximumNumber without fmaxf's sNaN canonicalization chatter.
__device__ __forceinline__ float vmaxf(float a, float b) {
  float r;
  asm("v_max_num_f32 %0, %1, %2" : "=v"(r) : "v"(a), "v"(b));
  return r;
}

// 16-lane max reduction: 4 single-instruction steps, DPP fused into the max.
__device__ __forceinline__ float maxred16(float x) {
  float r;
  asm("v_max_num_f32 %0, %1, %1 quad_perm:[1,0,3,2] row_mask:0xf bank_mask:0xf"
      : "=v"(r) : "v"(x));
  asm("v_max_num_f32 %0, %1, %1 quad_perm:[2,3,0,1] row_mask:0xf bank_mask:0xf"
      : "=v"(x) : "v"(r));
  asm("v_max_num_f32 %0, %1, %1 row_half_mirror row_mask:0xf bank_mask:0xf"
      : "=v"(r) : "v"(x));
  asm("v_max_num_f32 %0, %1, %1 row_mirror row_mask:0xf bank_mask:0xf"
      : "=v"(x) : "v"(r));
  return x;
}

// Concatenate two contiguous 16B LDS chunks into a v16h fragment.
__device__ __forceinline__ v16h lds_frag(const _Float16* p) {
  v8h lo = *(const v8h*)p;
  v8h hi = *(const v8h*)(p + 8);
  return __builtin_shufflevector(lo, hi, 0, 1, 2, 3, 4, 5, 6, 7, 8, 9, 10, 11,
                                 12, 13, 14, 15);
}

__global__ __launch_bounds__(128) void fa_fwd_f16wmma(
    const float* __restrict__ q, const float* __restrict__ k,
    const float* __restrict__ v, float* __restrict__ out) {
  __shared__ alignas(16) _Float16 Ks[2][64][KSTRIDE];    // K tiles, row-major
  __shared__ alignas(16) _Float16 Vt[2][64][VSTRIDE];    // V tiles, transposed
  __shared__ alignas(16) _Float16 Pbuf[4][16][PSTRIDE];  // per-wave P staging

  const int tid  = threadIdx.x;
  const int lane = tid & 31;
  const int wid  = tid >> 5;
  const int g    = lane >> 4;   // half-wave group
  const int n    = lane & 15;   // N index / A-matrix row
  const int bh   = blockIdx.y;
  const int qbase = blockIdx.x * 64 + wid * 16;

  const size_t base = (size_t)bh * SEQ * HD;
  const float QSCALE = 0.125f * 1.44269504088896340736f;  // scale * log2(e)

  // Cooperative V staging split: 4 keys x 8 d's per thread.
  const int vq  = tid >> 3;  // 0..15 : keys 4vq..4vq+3
  const int vdc = tid & 7;   // 0..7  : d in [8*vdc, 8*vdc+8)

  // Stage one 64-key K/V tile (fp32 global -> f16 LDS, V transposed).
  auto stage = [&](int buf, int kb) {
    const float* ktile = k + base + (size_t)kb * HD;
    const float* vtile = v + base + (size_t)kb * HD;
#pragma unroll
    for (int j = 0; j < 4; ++j) {
      int flat = tid * 8 + j * 1024;
      int row = flat >> 6, col = flat & 63;
      v4f a0 = *(const v4f*)(ktile + flat);
      v4f a1 = *(const v4f*)(ktile + flat + 4);
      v8h hk;
#pragma unroll
      for (int i = 0; i < 4; ++i) {
        hk[i]     = (_Float16)a0[i];
        hk[4 + i] = (_Float16)a1[i];
      }
      *(v8h*)&Ks[buf][row][col] = hk;
    }
    {
      const float* va = vtile + (size_t)(4 * vq) * HD + vdc * 8;
      v4f k0a = *(const v4f*)(va);          v4f k0b = *(const v4f*)(va + 4);
      v4f k1a = *(const v4f*)(va + HD);     v4f k1b = *(const v4f*)(va + HD + 4);
      v4f k2a = *(const v4f*)(va + 2*HD);   v4f k2b = *(const v4f*)(va + 2*HD + 4);
      v4f k3a = *(const v4f*)(va + 3*HD);   v4f k3b = *(const v4f*)(va + 3*HD + 4);
#pragma unroll
      for (int i = 0; i < 4; ++i) {
        v4h pa_, pb_;
        pa_[0] = (_Float16)k0a[i]; pa_[1] = (_Float16)k1a[i];
        pa_[2] = (_Float16)k2a[i]; pa_[3] = (_Float16)k3a[i];
        pb_[0] = (_Float16)k0b[i]; pb_[1] = (_Float16)k1b[i];
        pb_[2] = (_Float16)k2b[i]; pb_[3] = (_Float16)k3b[i];
        *(v4h*)&Vt[buf][vdc * 8 + i][4 * vq]     = pa_;
        *(v4h*)&Vt[buf][vdc * 8 + 4 + i][4 * vq] = pb_;
      }
    }
  };

  // ---- Q A-fragments (16x32 f16 each), softmax scale folded in ----
  const float* qp = q + base + (size_t)(qbase + n) * HD;
  v16h aq0, aq1;
#pragma unroll
  for (int h = 0; h < 16; ++h) {
    int dl = g * 16 + h;
    aq0[h] = (_Float16)(qp[dl] * QSCALE);
    aq1[h] = (_Float16)(qp[32 + dl] * QSCALE);
  }

  // Constant all-ones B fragment for row-sum WMMAs.
  v16h bones;
#pragma unroll
  for (int h = 0; h < 16; ++h) bones[h] = (_Float16)1.0f;

  float rmax[8], rsum[8];
  v8f oacc[4];
  v8f vzero = {};
#pragma unroll
  for (int r = 0; r < 8; ++r) { rmax[r] = -3.0e38f; rsum[r] = 0.0f; }
#pragma unroll
  for (int d = 0; d < 4; ++d) oacc[d] = vzero;

  _Float16* Pw = &Pbuf[wid][0][0];

  stage(0, 0);  // prologue: tile 0 into buffer 0
  __syncthreads();

  for (int it = 0; it < NTILES; ++it) {
    const int cur = it & 1;
    const _Float16* Kb = &Ks[cur][0][0];
    const _Float16* Vb = &Vt[cur][0][0];

    // ---- stage tile it+1 into the alternate buffer (latency hidden) ----
    if (it + 1 < NTILES) {
      stage(cur ^ 1, (it + 1) * 64);
      if (it + 2 < NTILES)
        __builtin_prefetch(k + base + (size_t)(it + 2) * 64 * HD + tid * 32, 0, 0);
    }

    // ---- S = (Q*scale*log2e) @ K^T : four 16x16 C tiles over 64 keys ----
    v8f c0 = vzero, c1 = vzero, c2 = vzero, c3 = vzero;
    {
      v16h b;
      b = lds_frag(Kb + (n)      * KSTRIDE + g * 16);
      c0 = __builtin_amdgcn_wmma_f32_16x16x32_f16(false, aq0, false, b,
                                                  (short)0, c0, false, false);
      b = lds_frag(Kb + (n)      * KSTRIDE + 32 + g * 16);
      c0 = __builtin_amdgcn_wmma_f32_16x16x32_f16(false, aq1, false, b,
                                                  (short)0, c0, false, false);
      b = lds_frag(Kb + (16 + n) * KSTRIDE + g * 16);
      c1 = __builtin_amdgcn_wmma_f32_16x16x32_f16(false, aq0, false, b,
                                                  (short)0, c1, false, false);
      b = lds_frag(Kb + (16 + n) * KSTRIDE + 32 + g * 16);
      c1 = __builtin_amdgcn_wmma_f32_16x16x32_f16(false, aq1, false, b,
                                                  (short)0, c1, false, false);
      b = lds_frag(Kb + (32 + n) * KSTRIDE + g * 16);
      c2 = __builtin_amdgcn_wmma_f32_16x16x32_f16(false, aq0, false, b,
                                                  (short)0, c2, false, false);
      b = lds_frag(Kb + (32 + n) * KSTRIDE + 32 + g * 16);
      c2 = __builtin_amdgcn_wmma_f32_16x16x32_f16(false, aq1, false, b,
                                                  (short)0, c2, false, false);
      b = lds_frag(Kb + (48 + n) * KSTRIDE + g * 16);
      c3 = __builtin_amdgcn_wmma_f32_16x16x32_f16(false, aq0, false, b,
                                                  (short)0, c3, false, false);
      b = lds_frag(Kb + (48 + n) * KSTRIDE + 32 + g * 16);
      c3 = __builtin_amdgcn_wmma_f32_16x16x32_f16(false, aq1, false, b,
                                                  (short)0, c3, false, false);
    }

    // ---- online softmax over 64 columns; row = r + 8*g ----
    float corr[8];
    float csum = 0.0f;
#pragma unroll
    for (int r = 0; r < 8; ++r) {
      float s0 = c0[r], s1 = c1[r], s2 = c2[r], s3 = c3[r];
      float mx = maxred16(vmaxf(vmaxf(s0, s1), vmaxf(s2, s3)));
      float mnew = vmaxf(rmax[r], mx);
      float e0 = exp2f(s0 - mnew);
      float e1 = exp2f(s1 - mnew);
      float e2 = exp2f(s2 - mnew);
      float e3 = exp2f(s3 - mnew);
      float cr = exp2f(rmax[r] - mnew);
      rmax[r] = mnew;
      corr[r] = cr;
      csum += cr;
      int row = r + 8 * g;
      Pw[row * PSTRIDE + n]      = (_Float16)e0;
      Pw[row * PSTRIDE + 16 + n] = (_Float16)e1;
      Pw[row * PSTRIDE + 32 + n] = (_Float16)e2;
      Pw[row * PSTRIDE + 48 + n] = (_Float16)e3;
    }

    // Rescale O only when some row max grew (wave-uniform; EXEC stays ones).
    if (!__all(csum == 8.0f)) {
#pragma unroll
      for (int d = 0; d < 4; ++d)
#pragma unroll
        for (int r = 0; r < 8; ++r) oacc[d][r] *= corr[r];
    }

    // ---- reload P in A-matrix layout (wave-private LDS, DS in-order) ----
    v16h pa0, pa1;
    {
      v8h x0 = *(const v8h*)&Pw[n * PSTRIDE + g * 16];
      v8h x1 = *(const v8h*)&Pw[n * PSTRIDE + 8 + g * 16];
      v8h x2 = *(const v8h*)&Pw[n * PSTRIDE + 32 + g * 16];
      v8h x3 = *(const v8h*)&Pw[n * PSTRIDE + 40 + g * 16];
      pa0 = __builtin_shufflevector(x0, x1, 0, 1, 2, 3, 4, 5, 6, 7, 8, 9, 10,
                                    11, 12, 13, 14, 15);
      pa1 = __builtin_shufflevector(x2, x3, 0, 1, 2, 3, 4, 5, 6, 7, 8, 9, 10,
                                    11, 12, 13, 14, 15);
    }

    // ---- row-sum of f16 P via WMMA against all-ones B ----
    v8f sumc = __builtin_amdgcn_wmma_f32_16x16x32_f16(false, pa0, false, bones,
                                                      (short)0, vzero, false,
                                                      false);
    sumc = __builtin_amdgcn_wmma_f32_16x16x32_f16(false, pa1, false, bones,
                                                  (short)0, sumc, false, false);
#pragma unroll
    for (int r = 0; r < 8; ++r) rsum[r] = fmaf(rsum[r], corr[r], sumc[r]);

    // ---- O += P @ V over 4 d-column tiles x 2 key halves ----
#pragma unroll
    for (int d = 0; d < 4; ++d) {
      v16h bv = lds_frag(Vb + (d * 16 + n) * VSTRIDE + g * 16);   // keys 0..31
      oacc[d] = __builtin_amdgcn_wmma_f32_16x16x32_f16(false, pa0, false, bv,
                                                       (short)0, oacc[d],
                                                       false, false);
      bv = lds_frag(Vb + (d * 16 + n) * VSTRIDE + 32 + g * 16);   // keys 32..63
      oacc[d] = __builtin_amdgcn_wmma_f32_16x16x32_f16(false, pa1, false, bv,
                                                       (short)0, oacc[d],
                                                       false, false);
    }

    __syncthreads();  // staged tile it+1 visible; buffer `cur` free to reuse
  }

  // ---- normalize and store fp32 ----
  float* op = out + base + (size_t)qbase * HD;
#pragma unroll
  for (int r = 0; r < 8; ++r) {
    float inv = 1.0f / rsum[r];
    int row = r + 8 * g;
#pragma unroll
    for (int d = 0; d < 4; ++d)
      op[(size_t)row * HD + d * 16 + n] = oacc[d][r] * inv;
  }
}

extern "C" void kernel_launch(void* const* d_in, const int* in_sizes, int n_in,
                              void* d_out, int out_size, void* d_ws,
                              size_t ws_size, hipStream_t stream) {
  (void)in_sizes; (void)n_in; (void)out_size; (void)d_ws; (void)ws_size;
  const float* q = (const float*)d_in[0];
  const float* k = (const float*)d_in[1];
  const float* v = (const float*)d_in[2];
  float* out = (float*)d_out;

  dim3 grid(SEQ / 64, 8 * 16);  // 16 q-tiles x 128 (batch*heads)
  dim3 block(128);              // 4 waves, one 16-row Q tile each
  hipLaunchKernelGGL(fa_fwd_f16wmma, grid, block, 0, stream, q, k, v, out);
}
